// SpatioTemporalLocalTransformerEncoderLayer_71717363908969
// MI455X (gfx1250) — compile-verified, matrix operand order checked by reading, not run on hardware
//
#include <hip/hip_runtime.h>
#include <hip/hip_bf16.h>
#include <math.h>

typedef __attribute__((ext_vector_type(16))) __bf16        v16bf;
typedef __attribute__((ext_vector_type(8)))  float         v8f;
typedef __attribute__((ext_vector_type(8)))  unsigned int  v8u;
typedef __attribute__((ext_vector_type(4)))  unsigned int  u32x4;
typedef __attribute__((ext_vector_type(8)))  unsigned int  u32x8;
typedef __attribute__((ext_vector_type(4)))  int           i32x4;
typedef __attribute__((ext_vector_type(8)))  int           i32x8;

#define D_MODEL 256
#define N_HEAD  8
#define HD      32
#define SLEN    128
#define D_FF    1024
#define NEGMASK -1e9f

// ---------- helpers ----------

__device__ __forceinline__ unsigned short f2bf(float x) {
  unsigned int u = __builtin_bit_cast(unsigned int, x);
  u += 0x7FFFu + ((u >> 16) & 1u);          // round-to-nearest-even
  return (unsigned short)(u >> 16);
}

__device__ __forceinline__ v8f wmma_bf16(v16bf a, v16bf b, v8f c) {
  return __builtin_amdgcn_wmma_f32_16x16x32_bf16(
      /*neg_a=*/false, a, /*neg_b=*/false, b,
      /*c_mod=*/(short)0, c, /*reuse_a=*/false, /*reuse_b=*/false);
}

// K offset inside a 32-wide K slab for VGPR v and lane-half (CDNA5 16-bit A layout)
__device__ __forceinline__ int kofs(int v, int hi) {
  return ((v < 4) ? (2 * v) : (8 + 2 * v)) + (hi ? 8 : 0);
}

// Load a 16x32 bf16 fragment from row-major u16 storage (rows = M for A,
// rows = N for B when the source holds W (out x in), i.e. B = W^T).
__device__ __forceinline__ v16bf load_frag_rm(const unsigned short* base, int ld,
                                              int row0, int k0) {
  int lane = threadIdx.x & 31;
  int hi = lane >> 4;
  int r = row0 + (lane & 15);
  v8u f;
#pragma unroll
  for (int v = 0; v < 8; ++v) {
    int k = k0 + kofs(v, hi);
    f[v] = *reinterpret_cast<const unsigned int*>(&base[r * ld + k]);
  }
  return __builtin_bit_cast(v16bf, f);
}

// Load a 32x16 B fragment from storage that is [K][N] row-major (K strided).
__device__ __forceinline__ v16bf load_frag_ks(const unsigned short* base, int ldk,
                                              int n0, int k0) {
  int lane = threadIdx.x & 31;
  int hi = lane >> 4;
  int n = n0 + (lane & 15);
  v8u f;
#pragma unroll
  for (int v = 0; v < 8; ++v) {
    int k = k0 + kofs(v, hi);
    unsigned int lo = base[k * ldk + n];
    unsigned int h2 = base[(k + 1) * ldk + n];
    f[v] = lo | (h2 << 16);
  }
  return __builtin_bit_cast(v16bf, f);
}

// Store a 16x16 f32 accumulator tile (C layout) as bf16 row-major.
__device__ __forceinline__ void store_tile_bf16(unsigned short* base, int ld, v8f acc,
                                                int m0, int n0) {
  int lane = threadIdx.x & 31;
  int n = n0 + (lane & 15);
  int mb = m0 + ((lane >> 4) ? 8 : 0);
#pragma unroll
  for (int r = 0; r < 8; ++r) base[(mb + r) * ld + n] = f2bf(acc[r]);
}

// ---------- prep kernels ----------

__global__ void k_cvt_bf16(const float* __restrict__ s, unsigned short* __restrict__ d, int n) {
  int i = blockIdx.x * blockDim.x + threadIdx.x;
  if (i < n) d[i] = f2bf(s[i]);
}

__global__ void k_smask(const int* __restrict__ coords, float* __restrict__ m) {
  int i = blockIdx.x * blockDim.x + threadIdx.x;   // 128*128
  int a = i >> 7, b = i & 127;
  int dx = coords[2 * a]     - coords[2 * b];     if (dx < 0) dx = -dx;
  int dy = coords[2 * a + 1] - coords[2 * b + 1]; if (dy < 0) dy = -dy;
  int ch = dx > dy ? dx : dy;
  m[i] = (ch > 4) ? NEGMASK : 0.0f;
}

__global__ void k_addbias(const float* __restrict__ s, const float* __restrict__ bias,
                          float* __restrict__ d, long n) {
  long i = (long)blockIdx.x * blockDim.x + threadIdx.x;
  if (i < n) d[i] = s[i] + bias[i & (D_MODEL - 1)];
}

// ---------- fused attention (one workgroup per (group, head)) ----------
// mode 0: spatial (contiguous tokens, additive mask from smask)
// mode 1: temporal (token stride 128*256, window mask 0 <= i-j <= 3)
//
// LDS plan (128KB total):
//   phase 0 : sh_raw = raw f32 x-tile [128 x 256], loaded by the Tensor Data Mover
//   phase 1+: first 64KB  -> sh_xn (bf16 normalized activations)
//             second 64KB -> sh_q/sh_k/sh_v (24KB) + per-wave P/O staging (32KB)

__global__ __launch_bounds__(256) void k_attn(
    const float* __restrict__ src, float* __restrict__ dst,
    const unsigned short* __restrict__ Wqkv, const float* __restrict__ bqkv,
    const unsigned short* __restrict__ Wo,
    const float* __restrict__ g, const float* __restrict__ smask, int mode) {
  __shared__ float sh_raw[SLEN * D_MODEL];                            // 128KB
  unsigned short* sh_xn = (unsigned short*)sh_raw;                    // 64KB overlay
  unsigned short* sh_q  = (unsigned short*)sh_raw + SLEN * D_MODEL;   // +64KB
  unsigned short* sh_k  = sh_q + SLEN * HD;
  unsigned short* sh_v  = sh_k + SLEN * HD;
  unsigned short* sh_p  = sh_v + SLEN * HD;                           // 8 waves x 16x128

  const int tid  = threadIdx.x;
  const int lane = tid & 31;
  const int wave = tid >> 5;
  const int head = blockIdx.x;
  const int grp  = blockIdx.y;

  long base;
  int  tstride;
  if (mode == 0) {
    base = (long)grp * SLEN * D_MODEL;
    tstride = D_MODEL;
  } else {
    int b = grp >> 7, s = grp & 127;
    base = ((long)b * (SLEN * SLEN) + s) * D_MODEL;
    tstride = SLEN * D_MODEL;
  }

  // ---- phase 0a: TDM async tensor load of the x tile (f32) into LDS ----
  // 2D tile: 256 elements per row, 128 rows, row stride = tstride elements.
  if (wave == 0) {
    unsigned long long ga = (unsigned long long)(const void*)(src + base);
    unsigned int lds_off = (unsigned int)(unsigned long long)(void*)sh_raw;
    u32x4 g0;
    g0[0] = 1u;                                              // count=1 (valid user D#)
    g0[1] = lds_off;                                         // lds_addr (bytes)
    g0[2] = (unsigned int)ga;                                // global_addr[31:0]
    g0[3] = (unsigned int)((ga >> 32) & 0x01FFFFFFull)       // global_addr[56:32]
            | 0x80000000u;                                   // type=2 ("image")
    u32x8 w;
    w[0] = 0x00020000u;                         // data_size=2 (4B), no multicast/pad
    w[1] = ((unsigned int)D_MODEL & 0xFFFFu) << 16;          // tensor_dim0[15:0]
    w[2] = ((unsigned int)SLEN & 0xFFFFu) << 16;             // dim0 hi16=0 | tensor_dim1[15:0]
    w[3] = ((unsigned int)D_MODEL) << 16;                    // dim1 hi16=0 | tile_dim0=256
    w[4] = (unsigned int)SLEN;                               // tile_dim1=128 (tile_dim2=0)
    w[5] = (unsigned int)tstride;                            // tensor_dim0_stride[31:0]
    w[6] = 0u;                                               // stride hi | dim1_stride lo
    w[7] = 0u;
    i32x4 z4 = {};
#if defined(__clang_major__) && (__clang_major__ >= 23)
    i32x8 z8 = {};
    __builtin_amdgcn_tensor_load_to_lds(g0, __builtin_bit_cast(i32x8, w), z4, z4, z8, 0);
#else
    __builtin_amdgcn_tensor_load_to_lds(g0, __builtin_bit_cast(i32x8, w), z4, z4, 0);
#endif
    __builtin_amdgcn_s_wait_tensorcnt(0);
  }
  __syncthreads();

  // ---- phase 0b: RMSNorm from LDS; stage row in VGPRs, then overwrite raw ----
  {
    int row = tid >> 1, half = tid & 1;
    const float* xr = sh_raw + row * D_MODEL + half * 128;
    float vals[128];
    float ss = 0.f;
#pragma unroll
    for (int j = 0; j < 128; ++j) { vals[j] = xr[j]; ss += vals[j] * vals[j]; }
    ss += __shfl_xor(ss, 1, 32);
    float rstd = rsqrtf(ss * (1.0f / D_MODEL) + 1.1920929e-07f);
    __syncthreads();                 // all raw reads complete before overlay write
    const float* gg = g + half * 128;
#pragma unroll
    for (int j = 0; j < 128; ++j)
      sh_xn[row * D_MODEL + half * 128 + j] = f2bf(vals[j] * rstd * gg[j]);
  }
  __syncthreads();

  const int m0 = wave * 16;   // this wave's 16-row M-tile

  // ---- phase 1: QKV projection for this head (WMMA) ----
  {
    v16bf afr[8];
#pragma unroll
    for (int ks = 0; ks < 8; ++ks) afr[ks] = load_frag_rm(sh_xn, D_MODEL, m0, ks * 32);

    unsigned short* shdst[3] = {sh_q, sh_k, sh_v};
#pragma unroll
    for (int q = 0; q < 3; ++q) {
#pragma unroll
      for (int nt = 0; nt < 2; ++nt) {
        int nrow0 = q * D_MODEL + head * HD + nt * 16;
        v8f acc = {};
#pragma unroll
        for (int ks = 0; ks < 8; ++ks) {
          v16bf b = load_frag_rm(Wqkv, D_MODEL, nrow0, ks * 32);
          acc = wmma_bf16(afr[ks], b, acc);
        }
        float bias = bqkv[nrow0 + (lane & 15)];
#pragma unroll
        for (int r = 0; r < 8; ++r) acc[r] += bias;
        store_tile_bf16(shdst[q], HD, acc, m0, nt * 16);
      }
    }
  }
  __syncthreads();

  unsigned short* myp = sh_p + wave * (16 * SLEN);

  // ---- phase 2: scores (WMMA, K=32 single step) + mask + softmax -> P ----
  {
    v16bf aq = load_frag_rm(sh_q, HD, m0, 0);
    float vals[8][8];
    const int n_l = lane & 15;
    const int mb  = m0 + ((lane >> 4) ? 8 : 0);
#pragma unroll
    for (int nt = 0; nt < 8; ++nt) {
      v16bf bk = load_frag_rm(sh_k, HD, nt * 16, 0);
      v8f z = {};
      v8f st = wmma_bf16(aq, bk, z);
      int n = nt * 16 + n_l;
#pragma unroll
      for (int r = 0; r < 8; ++r) {
        int m = mb + r;
        float madd;
        if (mode == 0) madd = smask[m * SLEN + n];
        else { int dd = m - n; madd = (dd >= 0 && dd <= 3) ? 0.f : NEGMASK; }
        vals[nt][r] = st[r] * 0.17677669529663687f + madd;   // 1/sqrt(32)
      }
    }
#pragma unroll
    for (int r = 0; r < 8; ++r) {
      float mx = vals[0][r];
#pragma unroll
      for (int nt = 1; nt < 8; ++nt) mx = fmaxf(mx, vals[nt][r]);
#pragma unroll
      for (int off = 1; off < 16; off <<= 1) mx = fmaxf(mx, __shfl_xor(mx, off, 32));
      float sum = 0.f;
#pragma unroll
      for (int nt = 0; nt < 8; ++nt) {
        float p = __expf(vals[nt][r] - mx);
        vals[nt][r] = p; sum += p;
      }
#pragma unroll
      for (int off = 1; off < 16; off <<= 1) sum += __shfl_xor(sum, off, 32);
      float inv = 1.f / sum;
#pragma unroll
      for (int nt = 0; nt < 8; ++nt) vals[nt][r] *= inv;
    }
    int mloc = ((lane >> 4) ? 8 : 0);
#pragma unroll
    for (int nt = 0; nt < 8; ++nt)
#pragma unroll
      for (int r = 0; r < 8; ++r)
        myp[(mloc + r) * SLEN + nt * 16 + n_l] = f2bf(vals[nt][r]);
  }

  // ---- phase 3: O = P @ V_h (WMMA), then stash O bf16 into staging ----
  {
    v8f o0 = {}, o1 = {};
#pragma unroll
    for (int ks = 0; ks < 4; ++ks) {
      v16bf a  = load_frag_rm(myp, SLEN, 0, ks * 32);
      v16bf b0 = load_frag_ks(sh_v, HD, 0,  ks * 32);
      v16bf b1 = load_frag_ks(sh_v, HD, 16, ks * 32);
      o0 = wmma_bf16(a, b0, o0);
      o1 = wmma_bf16(a, b1, o1);
    }
    store_tile_bf16(myp, HD, o0, 0, 0);
    store_tile_bf16(myp, HD, o1, 0, 16);
  }

  // ---- phase 4: partial out-proj O_h @ Wo_h^T, atomic accumulate ----
  {
    v16bf ao = load_frag_rm(myp, HD, 0, 0);      // 16 x 32, exactly one K step
    const int n_l = lane & 15;
    const int mb  = m0 + ((lane >> 4) ? 8 : 0);
#pragma unroll
    for (int nt = 0; nt < 16; ++nt) {
      v16bf b = load_frag_rm(Wo, D_MODEL, nt * 16, head * HD);
      v8f z = {};
      v8f acc = wmma_bf16(ao, b, z);
      int n = nt * 16 + n_l;
#pragma unroll
      for (int r = 0; r < 8; ++r)
        atomicAdd(&dst[base + (long)(mb + r) * tstride + n], acc[r]);
    }
  }
}

// ---------- fused FFN: x = x + W2 @ gelu(W1 @ rmsnorm(x) + b1) + b2 ----------

__global__ __launch_bounds__(256) void k_ffn(
    float* __restrict__ x,
    const unsigned short* __restrict__ W1, const float* __restrict__ b1,
    const unsigned short* __restrict__ W2, const float* __restrict__ b2,
    const float* __restrict__ g) {
  __shared__ unsigned short sh_xn[SLEN * D_MODEL];  // 64KB
  __shared__ unsigned short sh_h[SLEN * D_MODEL];   // 64KB (dff chunk of 256)

  const int tid = threadIdx.x, lane = tid & 31, wave = tid >> 5;
  const long base = (long)blockIdx.x * SLEN * D_MODEL;

  {
    int row = tid >> 1, half = tid & 1;
    const float* xr = x + base + row * D_MODEL + half * 128;
    float ss = 0.f;
#pragma unroll 8
    for (int j = 0; j < 128; ++j) { float v = xr[j]; ss += v * v; }
    ss += __shfl_xor(ss, 1, 32);
    float rstd = rsqrtf(ss * (1.0f / D_MODEL) + 1.1920929e-07f);
    const float* gg = g + half * 128;
#pragma unroll 8
    for (int j = 0; j < 128; ++j)
      sh_xn[row * D_MODEL + half * 128 + j] = f2bf(xr[j] * rstd * gg[j]);
  }
  __syncthreads();

  const int m0 = wave * 16;
  v16bf afr[8];
#pragma unroll
  for (int ks = 0; ks < 8; ++ks) afr[ks] = load_frag_rm(sh_xn, D_MODEL, m0, ks * 32);

  v8f yac[16];
#pragma unroll
  for (int nt = 0; nt < 16; ++nt) yac[nt] = (v8f){};

  for (int c = 0; c < 4; ++c) {        // dff chunks of 256
    // GEMM1: H = xn @ W1_c^T, then bias + exact GELU -> sh_h (wave-private rows)
#pragma unroll
    for (int nt = 0; nt < 16; ++nt) {
      v8f acc = {};
#pragma unroll
      for (int ks = 0; ks < 8; ++ks) {
        v16bf b = load_frag_rm(W1 + (long)c * 256 * D_MODEL, D_MODEL, nt * 16, ks * 32);
        acc = wmma_bf16(afr[ks], b, acc);
      }
      int n = c * 256 + nt * 16 + (lane & 15);
      float bb = b1[n];
#pragma unroll
      for (int r = 0; r < 8; ++r) {
        float h = acc[r] + bb;
        acc[r] = 0.5f * h * (1.f + erff(h * 0.7071067811865475f));
      }
      store_tile_bf16(sh_h, D_MODEL, acc, m0, nt * 16);
    }
    // GEMM2: Y += gelu(H) @ W2_c^T
#pragma unroll
    for (int ks = 0; ks < 8; ++ks) {
      v16bf a = load_frag_rm(sh_h, D_MODEL, m0, ks * 32);
#pragma unroll
      for (int nt = 0; nt < 16; ++nt) {
        v16bf b = load_frag_rm(W2, D_FF, nt * 16, c * 256 + ks * 32);
        yac[nt] = wmma_bf16(a, b, yac[nt]);
      }
    }
  }

  const int n_l = lane & 15;
  const int mb  = m0 + ((lane >> 4) ? 8 : 0);
#pragma unroll
  for (int nt = 0; nt < 16; ++nt) {
    int n = nt * 16 + n_l;
    float bb = b2[n];
#pragma unroll
    for (int r = 0; r < 8; ++r) {
      long idx = base + (long)(mb + r) * D_MODEL + n;
      x[idx] = x[idx] + yac[nt][r] + bb;
    }
  }
}

// ---------- host launcher ----------

extern "C" void kernel_launch(void* const* d_in, const int* in_sizes, int n_in,
                              void* d_out, int out_size, void* d_ws, size_t ws_size,
                              hipStream_t stream) {
  const float* x      = (const float*)d_in[0];
  const int*   coords = (const int*)d_in[1];
  const float* Wqkv_s = (const float*)d_in[3];
  const float* bqkv_s = (const float*)d_in[4];
  const float* Wo_s   = (const float*)d_in[5];
  const float* bo_s   = (const float*)d_in[6];
  const float* Wqkv_t = (const float*)d_in[7];
  const float* bqkv_t = (const float*)d_in[8];
  const float* Wo_t   = (const float*)d_in[9];
  const float* bo_t   = (const float*)d_in[10];
  const float* g1     = (const float*)d_in[11];
  const float* g2     = (const float*)d_in[12];
  const float* g3     = (const float*)d_in[13];
  const float* W1     = (const float*)d_in[14];
  const float* b1     = (const float*)d_in[15];
  const float* W2     = (const float*)d_in[16];
  const float* b2     = (const float*)d_in[17];
  float* out = (float*)d_out;

  const long N = 4L * 16384 * 256;

  // workspace layout: x1 (f32, 64MB) | bf16 weights (2MB) | smask (64KB)
  float* x1 = (float*)d_ws;
  unsigned short* wp = (unsigned short*)((char*)d_ws + N * sizeof(float));
  unsigned short* bWqkv_s = wp; wp += 768 * 256;
  unsigned short* bWo_s   = wp; wp += 256 * 256;
  unsigned short* bWqkv_t = wp; wp += 768 * 256;
  unsigned short* bWo_t   = wp; wp += 256 * 256;
  unsigned short* bW1     = wp; wp += 1024 * 256;
  unsigned short* bW2     = wp; wp += 256 * 1024;
  float* smask = (float*)wp;

  // prep: bf16 weights + spatial mask
  k_cvt_bf16<<<(768 * 256 + 255) / 256, 256, 0, stream>>>(Wqkv_s, bWqkv_s, 768 * 256);
  k_cvt_bf16<<<(256 * 256 + 255) / 256, 256, 0, stream>>>(Wo_s,   bWo_s,   256 * 256);
  k_cvt_bf16<<<(768 * 256 + 255) / 256, 256, 0, stream>>>(Wqkv_t, bWqkv_t, 768 * 256);
  k_cvt_bf16<<<(256 * 256 + 255) / 256, 256, 0, stream>>>(Wo_t,   bWo_t,   256 * 256);
  k_cvt_bf16<<<(1024 * 256 + 255) / 256, 256, 0, stream>>>(W1,    bW1,    1024 * 256);
  k_cvt_bf16<<<(256 * 1024 + 255) / 256, 256, 0, stream>>>(W2,    bW2,    256 * 1024);
  k_smask<<<64, 256, 0, stream>>>(coords, smask);

  const int nblk = (int)((N + 255) / 256);

  // spatial attention: x1 = x + bo_s + sum_h O_h Wo_h^T
  k_addbias<<<nblk, 256, 0, stream>>>(x, bo_s, x1, N);
  k_attn<<<dim3(N_HEAD, 512), 256, 0, stream>>>(x, x1, bWqkv_s, bqkv_s, bWo_s, g1, smask, 0);

  // temporal attention: out = x1 + bo_t + sum_h O_h Wo_h^T
  k_addbias<<<nblk, 256, 0, stream>>>(x1, bo_t, out, N);
  k_attn<<<dim3(N_HEAD, 512), 256, 0, stream>>>(x1, out, bWqkv_t, bqkv_t, bWo_t, g2, smask, 1);

  // FFN in-place on out
  k_ffn<<<512, 256, 0, stream>>>(out, bW1, b1, bW2, b2, g3);
}
